// MS_StandardConv_13228499271746
// MI455X (gfx1250) — matrix operand order repaired and checked by reading, not compiled, not run on hardware
//
#include <hip/hip_runtime.h>
#include <hip/hip_bf16.h>

typedef __attribute__((ext_vector_type(16))) _Float16 v16h;
typedef __attribute__((ext_vector_type(8)))  float    v8f;

#define CIN   64
#define COUT  128
#define HW    128
#define NB    16
#define PIX   (HW*HW)             // 16384
#define NBLK  (NB*HW*8)           // 16384 conv blocks (16 px each)
#define PACKW_HALVES (8*18*32*16) // 73728

// ---------------------------------------------------------------------------
// Pack W (OIHW f32) into a contiguous-k per-lane A-fragment layout (B matches) for
// V_WMMA_F32_16X16X32_F16:  packW[ct][chunk][lane][half], 16 halves = 32B/lane.
//   ct    = co tile (0..7), co = ct*16 + (lane&15)
//   chunk = tap*2 + cb  (tap = kh*3+kw, cb = ci half 0/1)
//   A half i, lane l: k = 16*(l>>4) + i
//   ci = cb*32 + k
// ---------------------------------------------------------------------------
__global__ void pack_weights(const float* __restrict__ W, _Float16* __restrict__ packW) {
    int j = blockIdx.x * 256 + threadIdx.x;
    if (j >= PACKW_HALVES) return;
    int i  =  j        & 15;
    int l  = (j >> 4)  & 31;
    int c  = (j >> 9)  % 18;
    int ct =  j / (18*32*16);
    int t  = c >> 1;
    int cb = (c & 1) * 32;
    int klo = (l >> 4) * 16;
    int k  = klo + i;
    int ci = cb + k;
    int co = ct*16 + (l & 15);
    int kh = t / 3, kw = t % 3;
    packW[j] = (_Float16)W[ ((co*CIN + ci)*3 + kh)*3 + kw ];
}

// ---------------------------------------------------------------------------
// Implicit-GEMM conv with WMMA + fused BN partial stats.
// One block = 16 output pixels (one row segment) x 128 output channels.
// 8 waves; each wave owns a 16-channel slice: 18 K-chunks of 32 -> 18 WMMAs.
// spike = round(clip(x,0,4)) computed on the fly into LDS (exact in f16).
// After the GEMM, each wave reduces its 16x16 D tile over the pixel lanes
// (__shfl_xor within 16-lane halves) and emits per-(channel,block) sum/sum^2.
// ---------------------------------------------------------------------------
__global__ void __launch_bounds__(256)
conv_wmma(const float* __restrict__ x, const _Float16* __restrict__ packW,
          float* __restrict__ out, float2* __restrict__ part) {
    __shared__ __align__(32) _Float16 patch[3*18*CIN]; // [kh][wcol(18)][ci(64)] = 6912B

    int blk = blockIdx.x;
    int wt  =  blk       & 7;      // w tile (16 px each)
    int h   = (blk >> 3) & 127;
    int n   =  blk >> 10;
    int w0  = wt * 16;
    int tid = threadIdx.x;

    // Fill patch with spike values; zero-pad outside the image.
    for (int i = tid; i < 3*18*CIN; i += 256) {
        int wcol =  i % 18;
        int rest =  i / 18;
        int ci   =  rest & 63;
        int kh   =  rest >> 6;
        int hh = h  + kh   - 1;
        int ww = w0 + wcol - 1;
        float v = 0.0f;
        if ((unsigned)hh < 128u && (unsigned)ww < 128u) {
            float xv = x[ ((n*CIN + ci)*HW + hh)*HW + ww ];
            v = rintf(fminf(fmaxf(xv, 0.0f), 4.0f));   // round-half-even == jnp.round
        }
        patch[(kh*18 + wcol)*CIN + ci] = (_Float16)v;
    }
    __syncthreads();

    int lane = tid & 31;
    int ct   = tid >> 5;          // wave id -> 16-channel slice
    int g    = lane >> 4;
    int nloc = lane & 15;         // = pixel (B/D column); A row = co_local

    // This wave's packed-A stream: v16h units, index (ct*18 + chunk)*32 + lane
    const v16h* pA = (const v16h*)packW + (size_t)(ct*18)*32 + lane;

    v8f acc = {};
    #pragma unroll
    for (int t = 0; t < 9; ++t) {
        int kh = t / 3, kw = t % 3;
        const _Float16* pB0 = &patch[ ((kh*18 + (nloc + kw))*CIN) + g*16 ];
        #pragma unroll
        for (int cb = 0; cb < 2; ++cb) {
            v16h a = pA[(t*2 + cb)*32];                       // 32B/lane global, coalesced
            v16h b = *(const v16h*)(pB0 + cb*32);             // 32B/lane LDS, contiguous
            acc = __builtin_amdgcn_wmma_f32_16x16x32_f16(
                      false, a, false, b, (short)0, acc, false, false);
        }
    }

    // D: lane l -> pixel = l&15 ; VGPR r -> co_local = 8*(l>>4)+r
    int coBase = ct*16 + g*8;
    #pragma unroll
    for (int r = 0; r < 8; ++r) {
        int co = coBase + r;
        out[ (((size_t)n*COUT + co)*HW + h)*HW + w0 + nloc ] = acc[r];
    }

    // Fused BN partials: reduce over the 16 pixel lanes of each half-wave.
    #pragma unroll
    for (int r = 0; r < 8; ++r) {
        float s  = acc[r];
        float s2 = s * s;
        #pragma unroll
        for (int m = 1; m <= 8; m <<= 1) {
            s  += __shfl_xor(s,  m, 32);
            s2 += __shfl_xor(s2, m, 32);
        }
        if (nloc == 0) {   // lanes 0 and 16 (different co via g in coBase)
            int co = coBase + r;
            part[(size_t)co*NBLK + blk] = make_float2(s, s2);  // [co][blk] for coalesced reduce
        }
    }
}

// ---------------------------------------------------------------------------
// Per-channel reduction of block partials -> fused BN scale/shift.
// One block per channel; contiguous float2 stream per channel; no atomics.
// ---------------------------------------------------------------------------
__global__ void bn_reduce(const float2* __restrict__ part,
                          const float* __restrict__ gamma,
                          const float* __restrict__ beta,
                          float* __restrict__ ss) {
    int c = blockIdx.x;
    int tid = threadIdx.x;
    const float2* p = part + (size_t)c * NBLK;
    float s = 0.f, s2 = 0.f;
    for (int b = tid; b < NBLK; b += 256) {
        float2 v = p[b];
        s += v.x; s2 += v.y;
    }
    __shared__ float r0[256], r1[256];
    r0[tid] = s; r1[tid] = s2;
    __syncthreads();
    for (int off = 128; off > 0; off >>= 1) {
        if (tid < off) { r0[tid] += r0[tid+off]; r1[tid] += r1[tid+off]; }
        __syncthreads();
    }
    if (tid == 0) {
        const float N = (float)NB * (float)PIX;
        float mean = r0[0] / N;
        float var  = r1[0] / N - mean*mean;
        float scale = gamma[c] * rsqrtf(var + 1e-5f);
        ss[2*c + 0] = scale;
        ss[2*c + 1] = beta[c] - mean*scale;
    }
}

__global__ void bn_apply(float* __restrict__ y, const float* __restrict__ ss) {
    size_t j   = (size_t)blockIdx.x * 256 + threadIdx.x;  // float4 index
    size_t idx = j * 4;
    int c = (int)((idx >> 14) & 127);                     // PIX = 16384 per (n,c)
    float scale = ss[2*c], shift = ss[2*c + 1];
    float4 v = ((float4*)y)[j];
    v.x = v.x*scale + shift;
    v.y = v.y*scale + shift;
    v.z = v.z*scale + shift;
    v.w = v.w*scale + shift;
    ((float4*)y)[j] = v;
}

// ---------------------------------------------------------------------------
extern "C" void kernel_launch(void* const* d_in, const int* in_sizes, int n_in,
                              void* d_out, int out_size, void* d_ws, size_t ws_size,
                              hipStream_t stream) {
    const float* x     = (const float*)d_in[0];   // [1,16,64,128,128]
    const float* W     = (const float*)d_in[1];   // [128,64,3,3]
    const float* gamma = (const float*)d_in[2];   // [128]
    const float* beta  = (const float*)d_in[3];   // [128]
    float* out = (float*)d_out;                   // [1,16,128,128,128]

    _Float16* packW = (_Float16*)d_ws;                                   //   147456 B
    float2*   part  = (float2*)((char*)d_ws + 147456);                   // 16777216 B
    float*    ss    = (float*)((char*)d_ws + 147456 + (size_t)NBLK*COUT*8); // 1024 B

    pack_weights<<<PACKW_HALVES/256, 256, 0, stream>>>(W, packW);
    conv_wmma   <<<NBLK,             256, 0, stream>>>(x, packW, out, part);
    bn_reduce   <<<COUT,             256, 0, stream>>>(part, gamma, beta, ss);
    bn_apply    <<<(NB*COUT*PIX)/4/256, 256, 0, stream>>>(out, ss);
}
